// SuperPoint_52828097741334
// MI455X (gfx1250) — compile-verified, matrix operand order checked by reading, not run on hardware
//
#include <hip/hip_runtime.h>
#include <hip/hip_bf16.h>
#include <stdint.h>

typedef __attribute__((ext_vector_type(16))) _Float16     v16h;
typedef __attribute__((ext_vector_type(8)))  float        v8f;
typedef __attribute__((ext_vector_type(4)))  unsigned int u32x4;
typedef __attribute__((ext_vector_type(8)))  int          i32x8;
typedef __attribute__((ext_vector_type(4)))  int          i32x4;

#define BATCH    2
#define THR      0.005f
#define CAND_CAP 65536

// ------------------------------------------------------------------
// Weight transform: OIHW f32 -> f16 [CoutPad][K], K = ks*ks*Cin,
// K index = (kh*ks+kw)*Cin + ci  (tap-major, channel inner) so that an
// implicit-GEMM K chunk is contiguous per output channel.
// ------------------------------------------------------------------
__global__ void k_transform_w(const float* __restrict__ w, _Float16* __restrict__ wt,
                              int Cout, int Cin, int ks, int CoutPad) {
  int K = ks * ks * Cin;
  size_t idx = (size_t)blockIdx.x * blockDim.x + threadIdx.x;
  if (idx >= (size_t)CoutPad * K) return;
  int co  = (int)(idx / K);
  int k   = (int)(idx % K);
  int tap = k / Cin, ci = k % Cin;
  int kh  = tap / ks, kw = tap % ks;
  float v = 0.0f;
  if (co < Cout) v = w[(((size_t)co * Cin + ci) * ks + kh) * ks + kw];
  wt[idx] = (_Float16)v;
}

// ------------------------------------------------------------------
// Implicit-GEMM conv (SAME padding) via WMMA f16->f32, specialized on
// (KS, CIN) so the tap x K-chunk loop fully unrolls into a dense WMMA
// stream. One wave = 32 consecutive pixels (two 16-pixel M tiles with
// independent accumulators) x 16 output channels: each LDS B fragment
// feeds two WMMAs (2x XDL ILP, 2x flop per global byte of A traffic).
// Block = 128 threads = 4 waves sharing one LDS weight tile (16 x K)
// staged once by the Tensor Data Mover.
// A fragment: lane m=l&15 -> pixel row, (l>>4)*16 -> K-half (16 halves).
// B fragment: lane n=l&15 -> out channel col, same K-half from LDS.
// ------------------------------------------------------------------
template<int KS, int CIN>
__global__ void __launch_bounds__(128)
k_conv_wmma(const _Float16* __restrict__ act, const _Float16* __restrict__ wt,
            const float* __restrict__ bias, _Float16* __restrict__ out,
            int H, int W, int Cout, int relu) {
  constexpr int K = KS * KS * CIN;
  __shared__ __align__(32) _Float16 lds_w[16 * K];
  int coutBase = blockIdx.y * 16;

#if defined(__gfx1250__) && __has_builtin(__builtin_amdgcn_tensor_load_to_lds)
  // ---- TDM: DMA the 16 x K f16 weight tile into LDS (2-D descriptor) ----
  if (threadIdx.x == 0) {
    unsigned lds_off = (unsigned)(uintptr_t)&lds_w[0];      // flat addr truncates to LDS offset
    unsigned long long ga = (unsigned long long)(uintptr_t)(wt + (size_t)coutBase * K);
    int CoutPadL = (int)gridDim.y * 16;
    u32x4 g0;
    g0[0] = 1u;                                             // count=1, user descriptor
    g0[1] = lds_off;                                        // LDS destination
    g0[2] = (unsigned)(ga & 0xFFFFFFFFu);                   // global_addr[31:0]
    g0[3] = (unsigned)((ga >> 32) & 0x01FFFFFFu) | (2u << 30); // addr[56:32] | type=2
    i32x8 g1;
    g1[0] = (1 << 16);                                      // data_size = 2 bytes
    g1[1] = (int)((unsigned)(K & 0xFFFF) << 16);            // tensor_dim0 lo16
    g1[2] = (int)(((unsigned)K >> 16) |
                  ((unsigned)(CoutPadL & 0xFFFF) << 16));   // dim0 hi16 | tensor_dim1 lo16
    g1[3] = (int)((unsigned)(K & 0xFFFF) << 16);            // dim1 hi16(0) | tile_dim0 = K
    g1[4] = 16;                                             // tile_dim1 = 16 rows, tile_dim2 = 0
    g1[5] = K;                                              // tensor_dim0_stride lo32
    g1[6] = 0;                                              // stride hi16 | dim1_stride lo16
    g1[7] = 0;
    i32x4 gz4 = {};                                         // groups 2/3 unused (2-D tensor)
    i32x8 gz8 = {};                                         // trailing group (clang-23 6-arg form)
    __builtin_amdgcn_tensor_load_to_lds(g0, g1, gz4, gz4, gz8, 0);
  }
  __builtin_amdgcn_s_wait_tensorcnt(0);
  __syncthreads();
#else
  for (int i = threadIdx.x; i < 16 * K; i += 128)
    lds_w[i] = wt[(size_t)(coutBase + i / K) * K + (i % K)];
  __syncthreads();
#endif

  int wave = (int)(blockIdx.x * 4 + (threadIdx.x >> 5));
  int lane = (int)(threadIdx.x & 31);
  int tilesPerRow = W >> 5;              // 32-pixel macro tile per wave
  int x0  = (wave % tilesPerRow) << 5;
  int row = wave / tilesPerRow;          // b*H + y
  int y   = row % H;
  int m     = lane & 15;
  int khalf = (lane >> 4) << 4;          // 0 or 16 (K-half per lane group)
  constexpr int pad = KS >> 1;

  v8f acc0 = {};
  v8f acc1 = {};
  const _Float16* brow = lds_w + (size_t)(lane & 15) * K + khalf;

#pragma unroll
  for (int kh = 0; kh < KS; ++kh) {
    int ys = y + kh - pad;
    bool yok = (ys >= 0) && (ys < H);
    int rowbase = row - y + (yok ? ys : 0);                 // b*H + ys (clamped)
#pragma unroll
    for (int kw = 0; kw < KS; ++kw) {
      int xs0 = x0 + m + kw - pad;
      int xs1 = xs0 + 16;                                   // >= 15, never < 0
      bool ok0 = yok && (xs0 >= 0) && (xs0 < W);
      bool ok1 = yok && (xs1 < W);
      const _Float16* ap0 = act + ((size_t)rowbase * W + (size_t)(ok0 ? xs0 : 0)) * CIN + khalf;
      const _Float16* ap1 = act + ((size_t)rowbase * W + (size_t)(ok1 ? xs1 : 0)) * CIN + khalf;
      const _Float16* bp  = brow + (kh * KS + kw) * CIN;
#pragma unroll
      for (int kc = 0; kc < CIN; kc += 32) {
        v16h b = *(const v16h*)(bp + kc);                   // shared across both M tiles
        v16h a0 = {};
        if (ok0) a0 = *(const v16h*)(ap0 + kc);
        v16h a1 = {};
        if (ok1) a1 = *(const v16h*)(ap1 + kc);
        acc0 = __builtin_amdgcn_wmma_f32_16x16x32_f16(false, a0, false, b,
                                                      (short)0, acc0, false, false);
        acc1 = __builtin_amdgcn_wmma_f32_16x16x32_f16(false, a1, false, b,
                                                      (short)0, acc1, false, false);
      }
    }
  }

  // Epilogue: D layout — VGPR r, lane l -> pixel M = r + 8*(l>>4), cout N = l&15
  int n  = lane & 15;
  int co = coutBase + n;
  if (co < Cout) {
    float bv = bias[co];
    int mbase = (lane >> 4) << 3;
#pragma unroll
    for (int r = 0; r < 8; ++r) {
      float v0 = acc0[r] + bv;
      float v1 = acc1[r] + bv;
      if (relu) { v0 = fmaxf(v0, 0.0f); v1 = fmaxf(v1, 0.0f); }
      size_t rb = (size_t)row * W;
      out[(rb + (size_t)(x0 + mbase + r)) * Cout + co]      = (_Float16)v0;
      out[(rb + (size_t)(x0 + 16 + mbase + r)) * Cout + co] = (_Float16)v1;
    }
  }
}

// ------------------------------------------------------------------
// conv1a: Cin=1 (K=9 — too skinny for WMMA, memory-trivial). Scalar.
// images NCHW f32 -> NHWC f16 (64ch) with ReLU.
// ------------------------------------------------------------------
__global__ void k_conv1a(const float* __restrict__ img, const float* __restrict__ w,
                         const float* __restrict__ bias, _Float16* __restrict__ out,
                         int H, int W) {
  size_t idx = (size_t)blockIdx.x * blockDim.x + threadIdx.x;
  size_t total = (size_t)BATCH * H * W * 64;
  if (idx >= total) return;
  int co = (int)(idx & 63);
  size_t p = idx >> 6;
  int x = (int)(p % W);
  size_t r = p / W;                      // b*H + y
  int y = (int)(r % H);
  float acc = bias[co];
  for (int kh = 0; kh < 3; ++kh) {
    int ys = y + kh - 1; if (ys < 0 || ys >= H) continue;
    for (int kw = 0; kw < 3; ++kw) {
      int xs = x + kw - 1; if (xs < 0 || xs >= W) continue;
      acc += img[(r - y + ys) * W + xs] * w[co * 9 + kh * 3 + kw];
    }
  }
  out[p * 64 + co] = (_Float16)fmaxf(acc, 0.0f);
}

// ------------------------------------------------------------------
// 2x2 max pool, NHWC f16
// ------------------------------------------------------------------
__global__ void k_pool2(const _Float16* __restrict__ in, _Float16* __restrict__ out,
                        int Ho, int Wo, int C) {
  size_t idx = (size_t)blockIdx.x * blockDim.x + threadIdx.x;
  size_t total = (size_t)BATCH * Ho * Wo * C;
  if (idx >= total) return;
  int c = (int)(idx % C);
  size_t p = idx / C;
  int xo = (int)(p % Wo);
  size_t r = p / Wo;
  int yo = (int)(r % Ho);
  int b  = (int)(r / Ho);
  int Hi = Ho * 2, Wi = Wo * 2;
  size_t base = (((size_t)b * Hi + (size_t)yo * 2) * Wi + (size_t)xo * 2) * C + c;
  float a0 = (float)in[base];
  float a1 = (float)in[base + C];
  float a2 = (float)in[base + (size_t)Wi * C];
  float a3 = (float)in[base + (size_t)Wi * C + C];
  out[idx] = (_Float16)fmaxf(fmaxf(a0, a1), fmaxf(a2, a3));
}

// ------------------------------------------------------------------
// Softmax over 65 ch, drop dustbin, pixel-shuffle 8x8, apply mask.
// sh: NHWC (B,128,128,65) f16 -> S: (B,1024,1024) f32
// ------------------------------------------------------------------
__global__ void k_softmax_ps(const _Float16* __restrict__ sh, const int* __restrict__ mask,
                             float* __restrict__ S, int Hc, int Wc) {
  size_t idx = (size_t)blockIdx.x * blockDim.x + threadIdx.x;
  size_t total = (size_t)BATCH * Hc * Wc;
  if (idx >= total) return;
  int x = (int)(idx % Wc);
  size_t r = idx / Wc;
  int y = (int)(r % Hc);
  int b = (int)(r / Hc);
  const _Float16* p = sh + idx * 65;
  float v[65];
  float mx = -1e30f;
  for (int c = 0; c < 65; ++c) { v[c] = (float)p[c]; mx = fmaxf(mx, v[c]); }
  float s = 0.0f;
  for (int c = 0; c < 65; ++c) { v[c] = __expf(v[c] - mx); s += v[c]; }
  float inv = 1.0f / s;
  int Hf = Hc * 8, Wf = Wc * 8;
  for (int c = 0; c < 64; ++c) {
    int yy = y * 8 + (c >> 3), xx = x * 8 + (c & 7);
    size_t o = ((size_t)b * Hf + yy) * Wf + xx;
    float mk = (mask[o] != 0) ? 1.0f : 0.0f;
    S[o] = v[c] * inv * mk;
  }
}

// ------------------------------------------------------------------
// Separable 9-tap max dilation (NMS building block), f32 full-res
// ------------------------------------------------------------------
__global__ void k_rowmax9(const float* __restrict__ in, float* __restrict__ out, int H, int W) {
  size_t idx = (size_t)blockIdx.x * blockDim.x + threadIdx.x;
  if (idx >= (size_t)BATCH * H * W) return;
  int x = (int)(idx % W);
  size_t r = idx / W;
  float mv = -1e30f;
  for (int d = -4; d <= 4; ++d) {
    int xs = x + d;
    if (xs >= 0 && xs < W) mv = fmaxf(mv, in[r * W + xs]);
  }
  out[idx] = mv;
}

__global__ void k_colmax9(const float* __restrict__ in, float* __restrict__ out, int H, int W) {
  size_t idx = (size_t)blockIdx.x * blockDim.x + threadIdx.x;
  if (idx >= (size_t)BATCH * H * W) return;
  int x = (int)(idx % W);
  size_t r = idx / W;
  int y = (int)(r % H);
  float mv = -1e30f;
  for (int d = -4; d <= 4; ++d) {
    int ys = y + d;
    if (ys >= 0 && ys < H) mv = fmaxf(mv, in[(r - y + ys) * W + x]);
  }
  out[idx] = mv;
}

__global__ void k_eq(const float* a, const float* p, float* mm, size_t n) {
  size_t i = (size_t)blockIdx.x * blockDim.x + threadIdx.x;
  if (i < n) mm[i] = (a[i] == p[i]) ? 1.0f : 0.0f;
}
__global__ void k_supp(const float* pmask, const float* s, float* ss, size_t n) {
  size_t i = (size_t)blockIdx.x * blockDim.x + threadIdx.x;
  if (i < n) ss[i] = (pmask[i] > 0.0f) ? 0.0f : s[i];
}
__global__ void k_update(float* mm, const float* ss, const float* pss, const float* pmask, size_t n) {
  size_t i = (size_t)blockIdx.x * blockDim.x + threadIdx.x;
  if (i >= n) return;
  bool newmax = (ss[i] == pss[i]);
  bool supp   = (pmask[i] > 0.0f);
  mm[i] = ((mm[i] > 0.0f) || (newmax && !supp)) ? 1.0f : 0.0f;
}
__global__ void k_finalize(const float* mm, const float* S, float* cand, int H, int W) {
  size_t idx = (size_t)blockIdx.x * blockDim.x + threadIdx.x;
  if (idx >= (size_t)BATCH * H * W) return;
  int x = (int)(idx % W);
  size_t r = idx / W;
  int y = (int)(r % H);
  float v = (mm[idx] > 0.0f) ? S[idx] : 0.0f;
  bool border = (x >= 4) && (x < W - 4) && (y >= 4) && (y < H - 4);
  cand[idx] = (border && v > THR) ? v : -1.0f;
}

// ------------------------------------------------------------------
// Top-K: atomic compaction of candidates, then per-batch iterative
// argmax selection (block-wide reduction, K=1024 rounds).
// ------------------------------------------------------------------
__global__ void k_zero_counts(int* c) { if (threadIdx.x < BATCH) c[threadIdx.x] = 0; }

__global__ void k_compact(const float* __restrict__ cand, float* __restrict__ cs,
                          int* __restrict__ ci, int* __restrict__ counts, int H, int W) {
  size_t idx = (size_t)blockIdx.x * blockDim.x + threadIdx.x;
  size_t per = (size_t)H * W;
  if (idx >= (size_t)BATCH * per) return;
  int b = (int)(idx / per);
  size_t p = idx % per;
  float v = cand[idx];
  if (v > THR) {
    int pos = atomicAdd(&counts[b], 1);
    if (pos < CAND_CAP) {
      cs[(size_t)b * CAND_CAP + pos] = v;
      ci[(size_t)b * CAND_CAP + pos] = (int)p;
    }
  }
}

__global__ void k_select(float* __restrict__ cs, const int* __restrict__ ci,
                         const int* __restrict__ counts, float* __restrict__ kpts,
                         float* __restrict__ scr, float* __restrict__ validf,
                         int W, int K) {
  int b = blockIdx.x, t = threadIdx.x;
  __shared__ float sv[256];
  __shared__ int   si[256];
  int cnt = counts[b];
  if (cnt > CAND_CAP) cnt = CAND_CAP;
  for (int k = 0; k < K; ++k) {
    float best = -1e30f; int bi = -1;
    for (int i = t; i < cnt; i += 256) {
      float v = cs[(size_t)b * CAND_CAP + i];
      if (v > best) { best = v; bi = i; }
    }
    sv[t] = best; si[t] = bi;
    __syncthreads();
    for (int st = 128; st > 0; st >>= 1) {
      if (t < st && sv[t + st] > sv[t]) { sv[t] = sv[t + st]; si[t] = si[t + st]; }
      __syncthreads();
    }
    if (t == 0) {
      int sel = si[0]; float v = sv[0];
      bool ok = (sel >= 0) && (v > THR);
      int pidx = (sel >= 0) ? ci[(size_t)b * CAND_CAP + sel] : 0;
      kpts[((size_t)b * K + k) * 2 + 0] = (float)(pidx % W);
      kpts[((size_t)b * K + k) * 2 + 1] = (float)(pidx / W);
      scr[(size_t)b * K + k]    = ok ? v : 0.0f;
      validf[(size_t)b * K + k] = ok ? 1.0f : 0.0f;
      if (sel >= 0) cs[(size_t)b * CAND_CAP + sel] = -1e30f;
    }
    __syncthreads();
  }
}

// ------------------------------------------------------------------
// Descriptor map L2-normalize over C per pixel, NHWC f16
// ------------------------------------------------------------------
__global__ void k_norm_dmap(const _Float16* __restrict__ d, _Float16* __restrict__ dn,
                            int Hc, int Wc, int C) {
  size_t idx = (size_t)blockIdx.x * blockDim.x + threadIdx.x;
  if (idx >= (size_t)BATCH * Hc * Wc) return;
  const _Float16* p = d + idx * C;
  float s = 0.0f;
  for (int c = 0; c < C; ++c) { float v = (float)p[c]; s += v * v; }
  float inv = 1.0f / fmaxf(sqrtf(s), 1e-12f);
  for (int c = 0; c < C; ++c) dn[idx * C + c] = (_Float16)((float)p[c] * inv);
}

// ------------------------------------------------------------------
// Bilinear descriptor sampling at keypoints + L2 normalize + valid mask
// block = (k, b); 256 threads = one per descriptor dim
// ------------------------------------------------------------------
__global__ void k_sample(const _Float16* __restrict__ dn, const float* __restrict__ kpts,
                         const float* __restrict__ validf, float* __restrict__ desc,
                         int Hc, int Wc, int C, int K) {
  int k = blockIdx.x, b = blockIdx.y, t = threadIdx.x;
  float kx = kpts[((size_t)b * K + k) * 2 + 0];
  float ky = kpts[((size_t)b * K + k) * 2 + 1];
  const float s = 8.0f;
  float dx = (float)Wc * s - s * 0.5f - 0.5f;
  float dy = (float)Hc * s - s * 0.5f - 0.5f;
  float gx = (kx - s * 0.5f + 0.5f) / dx * 2.0f - 1.0f;
  float gy = (ky - s * 0.5f + 0.5f) / dy * 2.0f - 1.0f;
  float px = (gx + 1.0f) * 0.5f * (float)(Wc - 1);
  float py = (gy + 1.0f) * 0.5f * (float)(Hc - 1);
  float x0f = floorf(px), y0f = floorf(py);
  float wx = px - x0f, wy = py - y0f;
  int x0i = min(max((int)x0f, 0), Wc - 1);
  int x1i = min(x0i + 1, Wc - 1);
  int y0i = min(max((int)y0f, 0), Hc - 1);
  int y1i = min(y0i + 1, Hc - 1);
  size_t base = (size_t)b * Hc * Wc * C;
  float d00 = (float)dn[base + ((size_t)y0i * Wc + x0i) * C + t];
  float d01 = (float)dn[base + ((size_t)y0i * Wc + x1i) * C + t];
  float d10 = (float)dn[base + ((size_t)y1i * Wc + x0i) * C + t];
  float d11 = (float)dn[base + ((size_t)y1i * Wc + x1i) * C + t];
  float v = d00 * (1 - wx) * (1 - wy) + d01 * wx * (1 - wy) +
            d10 * (1 - wx) * wy       + d11 * wx * wy;
  __shared__ float red[256];
  red[t] = v * v;
  __syncthreads();
  for (int st = 128; st > 0; st >>= 1) {
    if (t < st) red[t] += red[t + st];
    __syncthreads();
  }
  float inv = 1.0f / fmaxf(sqrtf(red[0]), 1e-12f);
  float vf = validf[(size_t)b * K + k];
  desc[((size_t)b * K + k) * C + t] = v * inv * vf;
}

// ------------------------------------------------------------------
// Host-side conv wrapper: transform weights (f32 OIHW -> f16 tiled),
// then implicit-GEMM WMMA conv (template-specialized per layer shape).
// ------------------------------------------------------------------
template<int KS, int CIN>
static void run_conv(const float* w, const float* bias, const _Float16* in, _Float16* out,
                     _Float16* wt, int H, int W, int Cout, int relu, hipStream_t s) {
  int CoutPad = (Cout + 15) & ~15;
  constexpr int K = KS * KS * CIN;
  size_t total = (size_t)CoutPad * K;
  k_transform_w<<<(unsigned)((total + 255) / 256), 256, 0, s>>>(w, wt, Cout, CIN, KS, CoutPad);
  size_t waves = (size_t)BATCH * H * (W / 32);       // 32-pixel macro tile per wave
  dim3 grid((unsigned)(waves / 4), (unsigned)(CoutPad / 16));
  k_conv_wmma<KS, CIN><<<grid, 128, 0, s>>>(in, wt, bias, out, H, W, Cout, relu);
}

static inline unsigned nblk(size_t n) { return (unsigned)((n + 255) / 256); }

extern "C" void kernel_launch(void* const* d_in, const int* in_sizes, int n_in,
                              void* d_out, int out_size, void* d_ws, size_t ws_size,
                              hipStream_t stream) {
  const float* images = (const float*)d_in[0];
  const int*   masks  = (const int*)d_in[1];
  const float* Wp[12]; const float* Bp[12];
  for (int i = 0; i < 12; ++i) { Wp[i] = (const float*)d_in[2 + 2 * i]; Bp[i] = (const float*)d_in[3 + 2 * i]; }
  // 0:conv1a 1:conv1b 2:conv2a 3:conv2b 4:conv3a 5:conv3b 6:conv4a 7:conv4b
  // 8:convPa 9:convPb 10:convDa 11:convDb

  const int H = 1024, Wd = 1024, Kp = 1024;
  char* ws = (char*)d_ws;
  size_t off = 0;
  auto carve = [&](size_t bytes) -> char* {
    char* p = ws + off;
    off += (bytes + 255) & ~(size_t)255;
    return p;
  };
  _Float16* A0 = (_Float16*)carve((size_t)BATCH * H * Wd * 64 * 2);  // ping
  _Float16* A1 = (_Float16*)carve((size_t)BATCH * H * Wd * 64 * 2);  // pong
  _Float16* WT = (_Float16*)carve((size_t)1 << 20);                  // weight tile buffer
  _Float16* SH = (_Float16*)carve((size_t)BATCH * 128 * 128 * 65 * 2);
  size_t npx = (size_t)BATCH * H * Wd;
  float* S  = (float*)carve(npx * 4);
  float* T  = (float*)carve(npx * 4);
  float* P  = (float*)carve(npx * 4);
  float* P2 = (float*)carve(npx * 4);
  float* MM = (float*)carve(npx * 4);
  float* SS = (float*)carve(npx * 4);
  float* CS = (float*)carve((size_t)BATCH * CAND_CAP * 4);
  int*   CI = (int*)carve((size_t)BATCH * CAND_CAP * 4);
  int*   CN = (int*)carve(256);
  _Float16* DN = (_Float16*)carve((size_t)BATCH * 128 * 128 * 256 * 2);

  float* outKpts  = (float*)d_out;                                  // (B,K,2)
  float* outDesc  = outKpts + (size_t)BATCH * Kp * 2;               // (B,K,256)
  float* outScr   = outDesc + (size_t)BATCH * Kp * 256;             // (B,K,1)
  float* outValid = outScr + (size_t)BATCH * Kp;                    // (B,K)

  // ---------------- backbone ----------------
  k_conv1a<<<nblk((size_t)BATCH * H * Wd * 64), 256, 0, stream>>>(images, Wp[0], Bp[0], A0, H, Wd);
  run_conv<3, 64>(Wp[1], Bp[1], A0, A1, WT, 1024, 1024, 64, 1, stream);     // conv1b
  k_pool2<<<nblk((size_t)BATCH * 512 * 512 * 64), 256, 0, stream>>>(A1, A0, 512, 512, 64);
  run_conv<3, 64>(Wp[2], Bp[2], A0, A1, WT, 512, 512, 64, 1, stream);       // conv2a
  run_conv<3, 64>(Wp[3], Bp[3], A1, A0, WT, 512, 512, 64, 1, stream);       // conv2b
  k_pool2<<<nblk((size_t)BATCH * 256 * 256 * 64), 256, 0, stream>>>(A0, A1, 256, 256, 64);
  run_conv<3, 64>(Wp[4], Bp[4], A1, A0, WT, 256, 256, 128, 1, stream);      // conv3a
  run_conv<3, 128>(Wp[5], Bp[5], A0, A1, WT, 256, 256, 128, 1, stream);     // conv3b
  k_pool2<<<nblk((size_t)BATCH * 128 * 128 * 128), 256, 0, stream>>>(A1, A0, 128, 128, 128);
  run_conv<3, 128>(Wp[6], Bp[6], A0, A1, WT, 128, 128, 128, 1, stream);     // conv4a
  run_conv<3, 128>(Wp[7], Bp[7], A1, A0, WT, 128, 128, 128, 1, stream);     // conv4b -> x in A0

  // ---------------- score head ----------------
  run_conv<3, 128>(Wp[8], Bp[8], A0, A1, WT, 128, 128, 256, 1, stream);     // convPa
  run_conv<1, 256>(Wp[9], Bp[9], A1, SH, WT, 128, 128, 65, 0, stream);      // convPb
  k_softmax_ps<<<nblk((size_t)BATCH * 128 * 128), 256, 0, stream>>>(SH, masks, S, 128, 128);

  // ---------------- NMS (separable 9x9 dilations) ----------------
  unsigned gpx = nblk(npx);
  k_rowmax9<<<gpx, 256, 0, stream>>>(S, T, H, Wd);
  k_colmax9<<<gpx, 256, 0, stream>>>(T, P, H, Wd);
  k_eq<<<gpx, 256, 0, stream>>>(S, P, MM, npx);
  for (int it = 0; it < 2; ++it) {
    k_rowmax9<<<gpx, 256, 0, stream>>>(MM, T, H, Wd);
    k_colmax9<<<gpx, 256, 0, stream>>>(T, P, H, Wd);       // P = dilated max_mask
    k_supp<<<gpx, 256, 0, stream>>>(P, S, SS, npx);        // SS = suppressed scores
    k_rowmax9<<<gpx, 256, 0, stream>>>(SS, T, H, Wd);
    k_colmax9<<<gpx, 256, 0, stream>>>(T, P2, H, Wd);      // P2 = dilated SS
    k_update<<<gpx, 256, 0, stream>>>(MM, SS, P2, P, npx);
  }
  k_finalize<<<gpx, 256, 0, stream>>>(MM, S, SS, H, Wd);   // SS reused as candidate map

  // ---------------- top-K keypoints ----------------
  k_zero_counts<<<1, 32, 0, stream>>>(CN);
  k_compact<<<gpx, 256, 0, stream>>>(SS, CS, CI, CN, H, Wd);
  k_select<<<BATCH, 256, 0, stream>>>(CS, CI, CN, outKpts, outScr, outValid, Wd, Kp);

  // ---------------- descriptor head ----------------
  run_conv<3, 128>(Wp[10], Bp[10], A0, A1, WT, 128, 128, 256, 1, stream);   // convDa
  run_conv<1, 256>(Wp[11], Bp[11], A1, A0, WT, 128, 128, 256, 0, stream);   // convDb -> dmap in A0
  k_norm_dmap<<<nblk((size_t)BATCH * 128 * 128), 256, 0, stream>>>(A0, DN, 128, 128, 256);
  k_sample<<<dim3(Kp, BATCH), 256, 0, stream>>>(DN, outKpts, outValid, outDesc, 128, 128, 256, Kp);

  (void)in_sizes; (void)n_in; (void)out_size; (void)ws_size;
}
